// modelRNN_52475910422665
// MI455X (gfx1250) — compile-verified, matrix operand order checked by reading, not run on hardware
//
#include <hip/hip_runtime.h>
#include <stdint.h>

typedef unsigned short u16;
typedef __attribute__((ext_vector_type(16))) __bf16 v16bf;
typedef __attribute__((ext_vector_type(8)))  float  v8f;

static constexpr int Bsz  = 256;   // batch
static constexpr int Ssz  = 128;   // seq len
static constexpr int Fsz  = 128;   // features (layer-0 input)
static constexpr int Hsz  = 400;   // hidden
static constexpr int KPAD = 416;   // 400 padded to 13*32 for wmma K-depth
static constexpr int N4   = 1600;  // 4*H gate width
static constexpr int BS   = Bsz * Ssz;
static constexpr int NBLK = (Bsz / 16) * (Hsz / 16); // 400 persistent tiles

// ---------- small helpers ----------
__device__ __forceinline__ u16 f2bf(float x) {
  union { float f; uint32_t u; } v; v.f = x;
  uint32_t u = v.u;
  return (u16)((u + 0x7FFFu + ((u >> 16) & 1u)) >> 16);  // RNE
}
__device__ __forceinline__ float bf2f(u16 h) {
  union { float f; uint32_t u; } v; v.u = ((uint32_t)h) << 16; return v.f;
}
__device__ __forceinline__ float sigmf(float x) { return 1.f / (1.f + __expf(-x)); }

union FragU { v16bf v; uint4 q[2]; };

// bf16 A/B fragment load, 16x32 tile, rows r0..r0+15, k-slab k0..k0+31.
// Lane layout per ISA: lanes 0-15 -> K=k0..k0+7 & k0+16..23; lanes 16-31 -> +8.
__device__ __forceinline__ v16bf load_frag_g(const u16* __restrict__ base,
                                             size_t rstride, int r0, int k0) {
  const int lane = threadIdx.x & 31;
  const u16* p = base + (size_t)(r0 + (lane & 15)) * rstride + (k0 + ((lane >> 4) << 3));
  FragU f;
  f.q[0] = *reinterpret_cast<const uint4*>(p);
  f.q[1] = *reinterpret_cast<const uint4*>(p + 16);
  return f.v;
}

__device__ __forceinline__ v16bf load_frag_lds(const u16* p0, int rstride, int k0) {
  const int lane = threadIdx.x & 31;
  const u16* p = p0 + (lane & 15) * rstride + k0 + ((lane >> 4) << 3);
  FragU f;
  f.q[0] = *reinterpret_cast<const uint4*>(p);
  f.q[1] = *reinterpret_cast<const uint4*>(p + 16);
  return f.v;
}

__device__ __forceinline__ v8f wmma_bf16(v16bf a, v16bf b, v8f c) {
  return __builtin_amdgcn_wmma_f32_16x16x32_bf16(false, a, false, b, (short)0, c,
                                                 false, false);
}

// ---------- conversion / init kernels ----------
__global__ void f32_to_bf16_kernel(const float* __restrict__ src,
                                   u16* __restrict__ dst, int n) {
  int i = blockIdx.x * blockDim.x + threadIdx.x;
  if (i < n) dst[i] = f2bf(src[i]);
}

// Build WT[n][k] = K[row_off + k][n] (bf16, zero-padded to kpad rows of K)
__global__ void build_wt_kernel(const float* __restrict__ K, u16* __restrict__ WT,
                                int row_off, int rows_valid, int kpad) {
  int i = blockIdx.x * blockDim.x + threadIdx.x;
  if (i >= N4 * kpad) return;
  int n = i / kpad, k = i - n * kpad;
  float v = (k < rows_valid) ? K[(size_t)(row_off + k) * N4 + n] : 0.f;
  WT[i] = f2bf(v);
}

// Zero pad columns H..KPAD-1 of an activation buffer [BS, KPAD]
__global__ void zero_pad_kernel(u16* __restrict__ Hall) {
  int i = blockIdx.x * blockDim.x + threadIdx.x;
  if (i >= BS * (KPAD - Hsz)) return;
  int row = i >> 4, c = i & 15;  // KPAD-Hsz == 16
  Hall[(size_t)row * KPAD + Hsz + c] = 0;
}

__global__ void reset_cnt_kernel(int* c) { *c = 0; }

// ---------- input-projection GEMM: Z[M,1600] = A[M,kpad] @ WT^T + bias ----------
// wave tile 32(M) x 64(N); grid (M/32, N4/64)
__global__ void __launch_bounds__(32)
gemm_zx_kernel(const u16* __restrict__ A, const u16* __restrict__ WT,
               const float* __restrict__ bias, float* __restrict__ Z, int kpad) {
  const int m0 = blockIdx.x * 32;
  const int n0 = blockIdx.y * 64;
  const int lane = threadIdx.x & 31;
  v8f acc[2][4] = {};
  for (int k = 0; k < kpad; k += 32) {
    v16bf a0 = load_frag_g(A, kpad, m0, k);
    v16bf a1 = load_frag_g(A, kpad, m0 + 16, k);
    v16bf b0 = load_frag_g(WT, kpad, n0, k);
    v16bf b1 = load_frag_g(WT, kpad, n0 + 16, k);
    v16bf b2 = load_frag_g(WT, kpad, n0 + 32, k);
    v16bf b3 = load_frag_g(WT, kpad, n0 + 48, k);
    acc[0][0] = wmma_bf16(a0, b0, acc[0][0]);
    acc[0][1] = wmma_bf16(a0, b1, acc[0][1]);
    acc[0][2] = wmma_bf16(a0, b2, acc[0][2]);
    acc[0][3] = wmma_bf16(a0, b3, acc[0][3]);
    acc[1][0] = wmma_bf16(a1, b0, acc[1][0]);
    acc[1][1] = wmma_bf16(a1, b1, acc[1][1]);
    acc[1][2] = wmma_bf16(a1, b2, acc[1][2]);
    acc[1][3] = wmma_bf16(a1, b3, acc[1][3]);
  }
#pragma unroll
  for (int mi = 0; mi < 2; ++mi)
#pragma unroll
    for (int ni = 0; ni < 4; ++ni) {
      int col = n0 + ni * 16 + (lane & 15);
      float bv = bias[col];
#pragma unroll
      for (int e = 0; e < 8; ++e) {
        int row = m0 + mi * 16 + e + ((lane >> 4) << 3);
        Z[(size_t)row * N4 + col] = acc[mi][ni][e] + bv;
      }
    }
}

// ---------- device-wide monotonic barrier (persistent kernel) ----------
__device__ __forceinline__ void grid_sync(int* cnt, int step) {
  __threadfence();
  if ((threadIdx.x & 31) == 0) {
    atomicAdd(cnt, 1);
    const int target = NBLK * (step + 1);
    while (atomicAdd(cnt, 0) < target) { __builtin_amdgcn_s_sleep(2); }
  }
  __threadfence();
}

// ---------- persistent LSTM layer ----------
// 400 workgroups x 1 wave; each owns a 16(batch) x 16(hidden) tile, all 4 gates.
// Wh tiles staged in LDS once; cell state in registers for all 128 steps.
__global__ void __launch_bounds__(32)
lstm_layer_kernel(const float* __restrict__ Zx,    // [BS,1600], bias included
                  const u16* __restrict__ WhT,     // [1600,KPAD] bf16
                  u16* __restrict__ Hall,          // [BS,KPAD] bf16 out (pads pre-zeroed)
                  int* __restrict__ sync_cnt) {
  __shared__ u16 lds_w[4][16][KPAD];  // 53,248 bytes
  const int m0 = (blockIdx.x % (Bsz / 16)) * 16;
  const int n0 = (blockIdx.x / (Bsz / 16)) * 16;
  const int lane = threadIdx.x & 31;

  // Stage Wh slice (gate g, cols n0..n0+15, full K) into LDS
  for (int g = 0; g < 4; ++g)
    for (int r = 0; r < 16; ++r) {
      const u16* src = WhT + (size_t)(n0 + g * Hsz + r) * KPAD;
      for (int kk = lane * 8; kk < KPAD; kk += 256)
        *reinterpret_cast<uint4*>(&lds_w[g][r][kk]) =
            *reinterpret_cast<const uint4*>(&src[kk]);
    }
  __syncthreads();

  float c_reg[8];
#pragma unroll
  for (int e = 0; e < 8; ++e) c_reg[e] = 0.f;

  const int nh = n0 + (lane & 15);
  for (int t = 0; t < Ssz; ++t) {
    v8f acc[4];
#pragma unroll
    for (int g = 0; g < 4; ++g) {
      int col = n0 + (lane & 15) + g * Hsz;
#pragma unroll
      for (int e = 0; e < 8; ++e) {
        int bb = m0 + e + ((lane >> 4) << 3);
        acc[g][e] = Zx[((size_t)bb * Ssz + t) * N4 + col];
      }
    }
    if (t > 0) {
      const u16* hbase = Hall + (size_t)(t - 1) * KPAD;  // row stride Ssz*KPAD
      for (int k = 0; k < KPAD; k += 32) {
        v16bf a = load_frag_g(hbase, (size_t)Ssz * KPAD, m0, k);
        acc[0] = wmma_bf16(a, load_frag_lds(&lds_w[0][0][0], KPAD, k), acc[0]);
        acc[1] = wmma_bf16(a, load_frag_lds(&lds_w[1][0][0], KPAD, k), acc[1]);
        acc[2] = wmma_bf16(a, load_frag_lds(&lds_w[2][0][0], KPAD, k), acc[2]);
        acc[3] = wmma_bf16(a, load_frag_lds(&lds_w[3][0][0], KPAD, k), acc[3]);
      }
    }
    // gates: i, j, f, o  (forget_bias = 1.0)
#pragma unroll
    for (int e = 0; e < 8; ++e) {
      int bb = m0 + e + ((lane >> 4) << 3);
      float gi = acc[0][e], gj = acc[1][e], gf = acc[2][e], go = acc[3][e];
      float cn = c_reg[e] * sigmf(gf + 1.f) + sigmf(gi) * tanhf(gj);
      c_reg[e] = cn;
      Hall[((size_t)bb * Ssz + t) * KPAD + nh] = f2bf(tanhf(cn) * sigmf(go));
    }
    grid_sync(sync_cnt, t);
  }
}

// ---------- output heads ----------
__global__ void heads_kernel(const u16* __restrict__ Hall,
                             const float* __restrict__ w_lg, const float* __restrict__ b_lg,
                             const float* __restrict__ w_sm2, const float* __restrict__ b_sm2,
                             const float* __restrict__ w5, const float* __restrict__ b5,
                             float* __restrict__ out) {
  int r = blockIdx.x * blockDim.x + threadIdx.x;
  if (r >= BS) return;
  const u16* h = Hall + (size_t)r * KPAD;
  float a[8] = {0, 0, 0, 0, 0, 0, 0, 0};
  for (int k = 0; k < Hsz; ++k) {
    float hv = bf2f(h[k]);
    a[0] += hv * w_lg[k];
    a[1] += hv * w_sm2[2 * k];
    a[2] += hv * w_sm2[2 * k + 1];
#pragma unroll
    for (int j = 0; j < 5; ++j) a[3 + j] += hv * w5[5 * k + j];
  }
  float* po = out + (size_t)r * 8;
  po[0] = sigmf(a[0] + b_lg[0]);
  float s0 = a[1] + b_sm2[0], s1 = a[2] + b_sm2[1];
  float m2 = fmaxf(s0, s1);
  float e0 = __expf(s0 - m2), e1 = __expf(s1 - m2);
  float inv2 = 1.f / (e0 + e1);
  po[1] = e0 * inv2;
  po[2] = e1 * inv2;
  float g[5], m5 = -3.4e38f;
#pragma unroll
  for (int j = 0; j < 5; ++j) { g[j] = a[3 + j] + b5[j]; m5 = fmaxf(m5, g[j]); }
  float s5 = 0.f;
#pragma unroll
  for (int j = 0; j < 5; ++j) { g[j] = __expf(g[j] - m5); s5 += g[j]; }
  float inv5 = 1.f / s5;
#pragma unroll
  for (int j = 0; j < 5; ++j) po[3 + j] = g[j] * inv5;
}

// ---------- host launch ----------
extern "C" void kernel_launch(void* const* d_in, const int* in_sizes, int n_in,
                              void* d_out, int out_size, void* d_ws, size_t ws_size,
                              hipStream_t stream) {
  (void)in_sizes; (void)n_in; (void)out_size; (void)ws_size;
  const float* x    = (const float*)d_in[0];
  const float* k0   = (const float*)d_in[1];
  const float* b0   = (const float*)d_in[2];
  const float* k1   = (const float*)d_in[3];
  const float* b1   = (const float*)d_in[4];
  const float* k2   = (const float*)d_in[5];
  const float* b2   = (const float*)d_in[6];
  const float* w_lg = (const float*)d_in[7];
  const float* blg  = (const float*)d_in[8];
  const float* wsm2 = (const float*)d_in[9];
  const float* bsm2 = (const float*)d_in[10];
  const float* w5   = (const float*)d_in[11];
  const float* b5   = (const float*)d_in[12];

  char* p = (char*)d_ws;
  auto take = [&](size_t bytes) {
    char* q = p; p += (bytes + 255) & ~(size_t)255; return q;
  };
  u16*   x_bf  = (u16*)take((size_t)BS * Fsz * 2);
  u16*   wxt0  = (u16*)take((size_t)N4 * Fsz * 2);
  u16*   wht0  = (u16*)take((size_t)N4 * KPAD * 2);
  u16*   wxt1  = (u16*)take((size_t)N4 * KPAD * 2);
  u16*   wht1  = (u16*)take((size_t)N4 * KPAD * 2);
  u16*   wxt2  = (u16*)take((size_t)N4 * KPAD * 2);
  u16*   wht2  = (u16*)take((size_t)N4 * KPAD * 2);
  float* Zx    = (float*)take((size_t)BS * N4 * 4);
  u16*   HallA = (u16*)take((size_t)BS * KPAD * 2);
  u16*   HallB = (u16*)take((size_t)BS * KPAD * 2);
  int*   cnt   = (int*)take(256);

  // weights / input to bf16 (transposed, K-padded)
  f32_to_bf16_kernel<<<(BS * Fsz + 255) / 256, 256, 0, stream>>>(x, x_bf, BS * Fsz);
  build_wt_kernel<<<(N4 * Fsz  + 255) / 256, 256, 0, stream>>>(k0, wxt0, 0,   Fsz, Fsz);
  build_wt_kernel<<<(N4 * KPAD + 255) / 256, 256, 0, stream>>>(k0, wht0, Fsz, Hsz, KPAD);
  build_wt_kernel<<<(N4 * KPAD + 255) / 256, 256, 0, stream>>>(k1, wxt1, 0,   Hsz, KPAD);
  build_wt_kernel<<<(N4 * KPAD + 255) / 256, 256, 0, stream>>>(k1, wht1, Hsz, Hsz, KPAD);
  build_wt_kernel<<<(N4 * KPAD + 255) / 256, 256, 0, stream>>>(k2, wxt2, 0,   Hsz, KPAD);
  build_wt_kernel<<<(N4 * KPAD + 255) / 256, 256, 0, stream>>>(k2, wht2, Hsz, Hsz, KPAD);

  struct Layer {
    const u16* Ain; int kin; const u16* wxt; const u16* wht;
    const float* bias; u16* Hout;
  } layers[3] = {
      {x_bf,  Fsz,  wxt0, wht0, b0, HallA},
      {HallA, KPAD, wxt1, wht1, b1, HallB},
      {HallB, KPAD, wxt2, wht2, b2, HallA},
  };

  for (int L = 0; L < 3; ++L) {
    zero_pad_kernel<<<(BS * 16 + 255) / 256, 256, 0, stream>>>(layers[L].Hout);
    reset_cnt_kernel<<<1, 1, 0, stream>>>(cnt);
    gemm_zx_kernel<<<dim3(BS / 32, N4 / 64), 32, 0, stream>>>(
        layers[L].Ain, layers[L].wxt, layers[L].bias, Zx, layers[L].kin);
    lstm_layer_kernel<<<NBLK, 32, 0, stream>>>(Zx, layers[L].wht, layers[L].Hout, cnt);
  }

  heads_kernel<<<(BS + 255) / 256, 256, 0, stream>>>(
      HallA, w_lg, blg, wsm2, bsm2, w5, b5, (float*)d_out);
}
